// OctreeConvBnRelu_24146306138635
// MI455X (gfx1250) — compile-verified, matrix operand order checked by reading, not run on hardware
//
#include <hip/hip_runtime.h>

typedef __attribute__((ext_vector_type(16))) __bf16 v16bf;
typedef __attribute__((ext_vector_type(8)))  __bf16 v8bf;
typedef __attribute__((ext_vector_type(8)))  float  v8f;
typedef __attribute__((ext_vector_type(4)))  int    v4i;

#define AS1 __attribute__((address_space(1)))
#define AS3 __attribute__((address_space(3)))

#define C_IN  32
#define C_OUT 32
#define BN_EPS 1e-5f
#define KMAX  27                      // reference kernel taps (3x3x3)
#define KC_MAX   (KMAX * C_IN)        // 864
#define KC_PAD   (KC_MAX + 8)         // 872 ushorts: bank-conflict-free LDS stride

#if __has_builtin(__builtin_amdgcn_global_load_async_to_lds_b128)
#define HAVE_ASYNC_LDS 1
#else
#define HAVE_ASYNC_LDS 0
#endif

// ---------- helpers ----------

// fp32 -> bf16, round-to-nearest-even, via bit ops
__device__ __forceinline__ unsigned short f2bf(float f) {
  unsigned int u = __float_as_uint(f);
  u += 0x7fffu + ((u >> 16) & 1u);
  return (unsigned short)(u >> 16);
}

// Load one lane's 16 bf16 elements of a 16x32 WMMA operand tile.
// Row is 32 contiguous bf16 (64 B). CDNA5 16-bit A layout:
//   lanes 0-15  (h=0): K 0..7  -> VGPR0-3,  K 16..23 -> VGPR4-7
//   lanes 16-31 (h=1): K 8..15 -> VGPR0-3,  K 24..31 -> VGPR4-7
// i.e. 16B chunks at byte offsets h*16 and 32+h*16. B mirrors A with N<->M.
__device__ __forceinline__ v16bf load_tile16(const unsigned short* base, int h) {
  const v8bf* p = reinterpret_cast<const v8bf*>(base);
  v8bf lo = p[h];
  v8bf hi = p[2 + h];
  v16bf r;
#pragma unroll
  for (int i = 0; i < 8; ++i) { r[i] = lo[i]; r[i + 8] = hi[i]; }
  return r;
}

// ---------- phase 1: x [C_IN, N] f32 -> xbf [N, C_IN] bf16 ----------
__global__ void xpose_bf16_kernel(const float* __restrict__ din,
                                  unsigned short* __restrict__ xbf, int Nn) {
  int n = blockIdx.x * blockDim.x + threadIdx.x;
  if (n >= Nn) return;
  __align__(16) unsigned short row[C_IN];
#pragma unroll
  for (int c = 0; c < C_IN; ++c) row[c] = f2bf(din[(size_t)c * Nn + n]);
  uint4* dst = reinterpret_cast<uint4*>(xbf + (size_t)n * C_IN);
  const uint4* src = reinterpret_cast<const uint4*>(row);
#pragma unroll
  for (int i = 0; i < (C_IN * 2) / 16; ++i) dst[i] = src[i];
}

// ---------- phase 2: W [KK, 32] f32 -> WT [32, KK] bf16 ; zero stats ----------
__global__ void wpose_bf16_kernel(const float* __restrict__ w,
                                  unsigned short* __restrict__ wt,
                                  float* __restrict__ stats, int KK) {
  int i = blockIdx.x * blockDim.x + threadIdx.x;
  if (i < 2 * C_OUT) stats[i] = 0.0f;
  int total = KK * C_OUT;
  if (i >= total) return;
  int kk = i >> 5;        // C_OUT == 32
  int n  = i & 31;
  wt[(size_t)n * KK + kk] = f2bf(w[i]);
}

// ---------- phase 3: gather + GEMM via v_wmma_f32_16x16x32_bf16 ----------
// 256 threads = 8 waves; wave handles 16 nodes x all 32 output channels.
// B (W^T) staged once per block into LDS via async-to-LDS; A gathered from
// global (L2-resident) so the VMEM pipe is dedicated to the irregular gather
// and the ds pipe feeds B.
template <int KT>
__global__ void __launch_bounds__(256, 2)
octconv_wmma_kernel(const unsigned short* __restrict__ xbf,   // [N][32] bf16
                    const int*            __restrict__ neigh, // [N][K]
                    const unsigned short* __restrict__ wt,    // [32][K*32] bf16 (W^T)
                    float*                __restrict__ out,   // [32][N] pre-BN
                    float*                __restrict__ stats, // [64] sums, sumsqs
                    int Nn, int Krt) {
  const int K  = KT > 0 ? KT : Krt;
  const int KC = K * C_IN;

  __shared__ unsigned short swt[C_OUT * KC_PAD];   // 55,808 B
  __shared__ float lstat[2 * C_OUT];

  const int tid  = threadIdx.x;
  const int wave = tid >> 5;
  const int lane = tid & 31;

  // ---- stage W^T into LDS (padded row stride), async when available ----
  {
    const int chunks = KC / 8;                     // 16B chunks per row
    for (int i = tid; i < C_OUT * chunks; i += 256) {
      int row = i / chunks;
      int ck  = i - row * chunks;
      const unsigned short* g = wt + (size_t)row * KC + ck * 8;
      unsigned short*       l = swt + row * KC_PAD + ck * 8;
#if HAVE_ASYNC_LDS
      __builtin_amdgcn_global_load_async_to_lds_b128(
          (AS1 v4i*)(void*)g, (AS3 v4i*)(void*)l, 0, 0);
#else
      *reinterpret_cast<uint4*>(l) = *reinterpret_cast<const uint4*>(g);
#endif
    }
  }
  if (tid < 2 * C_OUT) lstat[tid] = 0.0f;
#if HAVE_ASYNC_LDS
#if __has_builtin(__builtin_amdgcn_s_wait_asynccnt)
  __builtin_amdgcn_s_wait_asynccnt(0);
#else
  asm volatile("s_wait_asynccnt 0" ::: "memory");
#endif
#endif
  __syncthreads();

  const int node_base = (blockIdx.x * 8 + wave) * 16;
  if (node_base < Nn) {                       // wave-uniform: EXEC stays all-ones
    const int m = lane & 15;                  // node-in-tile / B column (channel)
    const int h = lane >> 4;                  // K-half select
    const unsigned short* srow0 = swt + m * KC_PAD;          // channels 0..15
    const unsigned short* srow1 = swt + (m + 16) * KC_PAD;   // channels 16..31
    const int* nrow = neigh + (size_t)(node_base + m) * K;

    v8f acc0 = {};
    v8f acc1 = {};
#pragma unroll
    for (int t = 0; t < K; ++t) {
      int idx = nrow[t];                                     // irregular gather
      v16bf a  = load_tile16(xbf + (size_t)idx * C_IN, h);   // A from global/L2
      v16bf b0 = load_tile16(srow0 + t * C_IN, h);           // B from LDS
      v16bf b1 = load_tile16(srow1 + t * C_IN, h);
      acc0 = __builtin_amdgcn_wmma_f32_16x16x32_bf16(false, a, false, b0,
                                                     (short)0, acc0, false, false);
      acc1 = __builtin_amdgcn_wmma_f32_16x16x32_bf16(false, a, false, b1,
                                                     (short)0, acc1, false, false);
    }

    // D layout: lane L holds channel (L&15)+16*tile, VGPR r -> node_base + 8*h + r
    // => each lane stores 8 consecutive nodes of one channel: one 32B store.
    const int mrow = node_base + 8 * h;
    *reinterpret_cast<v8f*>(out + (size_t)m * Nn + mrow)        = acc0;
    *reinterpret_cast<v8f*>(out + (size_t)(m + 16) * Nn + mrow) = acc1;

    // per-channel sum / sumsq over this wave's 16 nodes
    float s0 = 0.f, q0 = 0.f, s1 = 0.f, q1 = 0.f;
#pragma unroll
    for (int r = 0; r < 8; ++r) {
      s0 += acc0[r]; q0 += acc0[r] * acc0[r];
      s1 += acc1[r]; q1 += acc1[r] * acc1[r];
    }
    atomicAdd(&lstat[m],              s0);
    atomicAdd(&lstat[C_OUT + m],      q0);
    atomicAdd(&lstat[m + 16],         s1);
    atomicAdd(&lstat[C_OUT + m + 16], q1);
  }
  __syncthreads();
  if (tid < 2 * C_OUT) atomicAdd(&stats[tid], lstat[tid]);
}

// ---------- phase 4: BN (training stats) + ReLU, in place on [32][N] ----------
__global__ void bn_relu_kernel(float* __restrict__ out,
                               const float* __restrict__ stats,
                               const float* __restrict__ gamma,
                               const float* __restrict__ beta,
                               int Nn) {
  long long i = (long long)blockIdx.x * blockDim.x + threadIdx.x;
  long long total = (long long)C_OUT * Nn;
  if (i >= total) return;
  int c = (int)(i / Nn);
  float inv  = 1.0f / (float)Nn;
  float mean = stats[c] * inv;
  float var  = stats[C_OUT + c] * inv - mean * mean;
  float sc   = rsqrtf(var + BN_EPS) * gamma[c];
  float sh   = beta[c] - mean * sc;
  float v    = out[i] * sc + sh;
  out[i]     = v > 0.0f ? v : 0.0f;
}

// ---------- launcher ----------
extern "C" void kernel_launch(void* const* d_in, const int* in_sizes, int n_in,
                              void* d_out, int out_size, void* d_ws, size_t ws_size,
                              hipStream_t stream) {
  const float* din   = (const float*)d_in[0];  // [1, C_IN, N, 1]
  const int*   neigh = (const int*)  d_in[1];  // [N, K]
  const float* w     = (const float*)d_in[2];  // [K*C_IN, C_OUT]
  const float* gamma = (const float*)d_in[3];  // [C_OUT]
  const float* beta  = (const float*)d_in[4];  // [C_OUT]
  float* out = (float*)d_out;                  // [C_OUT, N] flat

  const int Nn = in_sizes[0] / C_IN;           // 131072
  const int K  = in_sizes[1] / Nn;             // 27
  const int KK = K * C_IN;                     // 864

  // workspace: xbf [N*32 u16] | WT [32*KK u16] | stats [64 f32]
  char* ws = (char*)d_ws;
  unsigned short* xbf = (unsigned short*)ws;
  size_t off = (size_t)Nn * C_IN * sizeof(unsigned short);
  off = (off + 255) & ~(size_t)255;
  unsigned short* wt = (unsigned short*)(ws + off);
  off += (size_t)C_OUT * KK * sizeof(unsigned short);
  off = (off + 255) & ~(size_t)255;
  float* stats = (float*)(ws + off);

  xpose_bf16_kernel<<<(Nn + 255) / 256, 256, 0, stream>>>(din, xbf, Nn);

  int wtot = KK * C_OUT;
  int wthreads = wtot > 2 * C_OUT ? wtot : 2 * C_OUT;
  wpose_bf16_kernel<<<(wthreads + 255) / 256, 256, 0, stream>>>(w, wt, stats, KK);

  int gblocks = (Nn + 127) / 128;
  if (K == KMAX)
    octconv_wmma_kernel<KMAX><<<gblocks, 256, 0, stream>>>(xbf, neigh, wt, out,
                                                           stats, Nn, K);
  else
    octconv_wmma_kernel<0><<<gblocks, 256, 0, stream>>>(xbf, neigh, wt, out,
                                                        stats, Nn, K);

  long long tot = (long long)C_OUT * Nn;
  bn_relu_kernel<<<(int)((tot + 255) / 256), 256, 0, stream>>>(out, stats, gamma,
                                                               beta, Nn);
}
